// Model_90194313216718
// MI455X (gfx1250) — matrix operand order repaired and checked by reading, not compiled, not run on hardware
//
#include <hip/hip_runtime.h>

// ---------------------------------------------------------------------------
// BiLSTM + (batch-axis) MHA + CRF Viterbi for MI455X (gfx1250, wave32, WMMA)
//
//   1. convert weights f32->bf16; fold out_w/lw into a single [17,512] matrix
//   2. embedding gather -> x_bf16 [B*S,256]
//   3. GEMM xg_{f,b} = x @ Wi^T + (bi+bh)   [65536,1024] f32
//   4. persistent-WG LSTM recurrence (1 WG/direction, h in LDS, c in VGPRs,
//      Wh streamed from L2 each step; 4 gate accumulators live per wave to
//      stay under the 1024-thread VGPR budget -> no scratch spills)
//   5. GEMM qkv = h @ in_w^T + in_b         [65536,1536] bf16
//   6. attention per (s,head): q/k staged into LDS by the Tensor Data Mover
//      (tensor_load_to_lds + s_wait_tensorcnt), WMMA scores + o
//   7. GEMM em = o @ Wem^T + bem            [65536,32] f32 (17 valid cols)
//   8. Viterbi forward (wave per batch row) + backtrack
// ---------------------------------------------------------------------------

typedef __bf16 bf16_t;
typedef __attribute__((ext_vector_type(16))) __bf16 v16bf;
typedef __attribute__((ext_vector_type(8)))  float  v8f;
typedef __attribute__((ext_vector_type(4)))  unsigned uint4v;
typedef __attribute__((ext_vector_type(8)))  int      int8v;
typedef __attribute__((ext_vector_type(4)))  int      int4v;

#define Bn   64
#define Sn   1024
#define Dn   256
#define Hn   256
#define En   512
#define Tn   17
#define NHn  4
#define HDn  128

static __device__ inline bf16_t f2bf(float f) {
  unsigned u = __builtin_bit_cast(unsigned, f);
  u += 0x7FFFu + ((u >> 16) & 1u);          // round-to-nearest-even
  unsigned short h = (unsigned short)(u >> 16);
  return __builtin_bit_cast(bf16_t, h);
}

static __device__ inline v8f zero8() {
  v8f z = {0.f, 0.f, 0.f, 0.f, 0.f, 0.f, 0.f, 0.f};
  return z;
}

static __device__ inline v8f wmma_bf16(v16bf a, v16bf b, v8f c) {
  return __builtin_amdgcn_wmma_f32_16x16x32_bf16(false, a, false, b,
                                                 (short)0, c, false, false);
}

// Opaque 0: prevents LICM from hoisting (and then spilling) loop-invariant
// weight loads out of the 1024-step recurrence loop.
static __device__ inline int opaque_zero() {
  int z;
  asm volatile("v_mov_b32 %0, 0" : "=v"(z));
  return z;
}

// A-matrix fragment: 16x32 bf16, row-major source [.,stride].
// Lanes 0-15 (M=lane) hold K 0-7 & 16-23; lanes 16-31 hold K 8-15 & 24-31.
static __device__ inline v16bf load_frag_a(const bf16_t* base, int stride,
                                           int m0, int k0, int lane) {
  int r = lane & 15, half = lane >> 4;
  const bf16_t* p = base + (size_t)(m0 + r) * stride + k0 + half * 8;
  v16bf f;
#pragma unroll
  for (int e = 0; e < 8; ++e) f[e] = p[e];
#pragma unroll
  for (int e = 0; e < 8; ++e) f[e + 8] = p[16 + e];
  return f;
}

// B-matrix fragment: 32x16 bf16 from [N,K] row-major source (weights).
// Lanes 0-15 (N=lane) hold K 0-7 & 16-23; lanes 16-31 hold K 8-15 & 24-31.
static __device__ inline v16bf load_frag_b(const bf16_t* base, int stride,
                                           int n0, int k0, int lane) {
  const bf16_t* p = base + (size_t)(n0 + (lane & 15)) * stride + k0 + (lane >> 4) * 8;
  v16bf f;
#pragma unroll
  for (int e = 0; e < 8; ++e) f[e] = p[e];
#pragma unroll
  for (int e = 0; e < 8; ++e) f[e + 8] = p[16 + e];
  return f;
}

// ---------------------------------------------------------------------------
// Tensor Data Mover: 2-D tile load global->LDS (D# per CDNA5 ISA ch.8).
// data_size = 4B; tile_w_dw dwords/row, tile_rows rows, row stride in dwords.
// Optional LDS padding: pad_iv_code (0=2..7=256 dwords interval),
// pad_amt_code (0=1..127=128 dwords inserted).
static __device__ inline void tdm_load_2d(const void* gaddr, unsigned lds_off,
                                          unsigned tile_w_dw, unsigned tile_rows,
                                          unsigned row_stride_dw,
                                          unsigned pad_iv_code, unsigned pad_amt_code,
                                          int pad_en) {
  unsigned long long ga = (unsigned long long)(size_t)gaddr;
  uint4v g0;
  g0[0] = 1u;                                      // count=1 (valid user D#)
  g0[1] = lds_off;                                 // lds_addr [63:32]
  g0[2] = (unsigned)(ga & 0xFFFFFFFFu);            // global_addr low
  g0[3] = (unsigned)((ga >> 32) & 0x1FFFFFFu)      // global_addr [120:96]
        | (2u << 30);                              // type=2 ("image") [127:126]
  int8v g1;
  unsigned d0 = (2u << 16)                         // data_size = 4 bytes
              | ((unsigned)(pad_en ? 1u : 0u) << 20)
              | (pad_iv_code << 22) | (pad_amt_code << 25);
  unsigned td0 = row_stride_dw;                    // tensor_dim0 (no x-OOB)
  unsigned td1 = 0x100000u;                        // tensor_dim1 (no y-OOB)
  g1[0] = (int)d0;
  g1[1] = (int)((td0 & 0xFFFFu) << 16);            // [63:48] tensor_dim0 lo16
  g1[2] = (int)(((td0 >> 16) & 0xFFFFu) | ((td1 & 0xFFFFu) << 16));
  g1[3] = (int)(((td1 >> 16) & 0xFFFFu) | ((tile_w_dw & 0xFFFFu) << 16)); // tile_dim0
  g1[4] = (int)(tile_rows & 0xFFFFu);              // tile_dim1; tile_dim2=0
  g1[5] = (int)row_stride_dw;                      // tensor_dim0_stride lo32
  g1[6] = 0;                                       // stride hi16 | dim1_stride lo16
  g1[7] = 0;
  int4v z4 = {0, 0, 0, 0};
  int8v z8 = {0, 0, 0, 0, 0, 0, 0, 0};
  __builtin_amdgcn_tensor_load_to_lds(g0, g1, z4, z4, z8, 0);
}

// ---------------------------------------------------------------------------
__global__ void f2bf_kernel(const float* __restrict__ in, bf16_t* __restrict__ out, int n) {
  int i = blockIdx.x * blockDim.x + threadIdx.x;
  if (i < n) out[i] = f2bf(in[i]);
}

// Fold out_w/lw into Wem [32,512] (rows 17..31 zero) and bem[32]
__global__ __launch_bounds__(512) void wem_kernel(
    const float* __restrict__ lw, const float* __restrict__ lb,
    const float* __restrict__ out_w, const float* __restrict__ out_b,
    bf16_t* __restrict__ wem, float* __restrict__ bem) {
  int t = blockIdx.x, e = threadIdx.x;
  float acc = 0.f;
  if (t < Tn) {
    for (int j = 0; j < En; ++j) acc += lw[t * En + j] * out_w[j * En + e];
  }
  wem[(size_t)t * En + e] = f2bf(acc);
  if (e == 0) {
    float bb = 0.f;
    if (t < Tn) {
      for (int j = 0; j < En; ++j) bb += lw[t * En + j] * out_b[j];
      bb += lb[t];
    }
    bem[t] = bb;
  }
}

__global__ __launch_bounds__(256) void embed_kernel(
    const int* __restrict__ ids, const float* __restrict__ embed,
    bf16_t* __restrict__ x) {
  int row = blockIdx.x, d = threadIdx.x;
  int id = ids[row];
  x[(size_t)row * Dn + d] = f2bf(embed[(size_t)id * Dn + d]);
}

// ---------------------------------------------------------------------------
// Generic bf16 WMMA GEMM: C[m,n] = sum_k A[m,k]*B[n,k] + bias0[n] + bias1[n]
template <bool OUT_BF16>
__global__ __launch_bounds__(256) void gemm_bf16_kernel(
    const bf16_t* __restrict__ A, const bf16_t* __restrict__ B,
    const float* __restrict__ bias0, const float* __restrict__ bias1,
    void* __restrict__ C, int M, int N, int K, int ldc) {
  int lane = threadIdx.x & 31;
  int wave = blockIdx.x * 8 + (threadIdx.x >> 5);
  int ntiles = N >> 4;
  int mt = wave / ntiles, nt = wave % ntiles;
  if (mt >= (M >> 4)) return;
  int m0 = mt << 4, n0 = nt << 4;
  v8f acc = zero8();
  for (int k0 = 0; k0 < K; k0 += 32) {
    v16bf af = load_frag_a(A, K, m0, k0, lane);
    v16bf bf = load_frag_b(B, K, n0, k0, lane);
    if (k0 + 32 < K) {
      __builtin_prefetch(A + (size_t)(m0 + (lane & 15)) * K + k0 + 32, 0, 1);
      __builtin_prefetch(B + (size_t)(n0 + (lane & 15)) * K + k0 + 32, 0, 1);
    }
    acc = wmma_bf16(af, bf, acc);
  }
  int col = n0 + (lane & 15);
  float bb = (bias0 ? bias0[col] : 0.f) + (bias1 ? bias1[col] : 0.f);
  int rbase = m0 + ((lane >> 4) << 3);
#pragma unroll
  for (int r = 0; r < 8; ++r) {
    float v = acc[r] + bb;
    size_t idx = (size_t)(rbase + r) * ldc + col;
    if (OUT_BF16) ((bf16_t*)C)[idx] = f2bf(v);
    else          ((float*)C)[idx]  = v;
  }
}

// ---------------------------------------------------------------------------
// Persistent LSTM recurrence. grid=2 (dir), block=1024 (32 waves).
// Wave w: h-col tile j=w&15, M half mh=w>>4 (2 M-tiles of 16 rows each).
// kc-outer / gate-inner loop keeps only 1 A-frag + 1 B-frag + 4 accumulators
// live (~100 VGPRs) so nothing spills at 8 waves/SIMD.
__global__ __launch_bounds__(1024) void lstm_kernel(
    const float* __restrict__ xg_f, const float* __restrict__ xg_b,
    const bf16_t* __restrict__ wh_f, const bf16_t* __restrict__ wh_b,
    bf16_t* __restrict__ h_out) {
  __shared__ bf16_t hbuf[Bn][264];   // h_{s-1}, bf16, 16B-aligned padded rows
  const int dir = blockIdx.x;
  const float*  xg = dir ? xg_b : xg_f;
  const bf16_t* Wh = dir ? wh_b : wh_f;
  const int colofs = dir ? Hn : 0;

  for (int i = threadIdx.x; i < Bn * 264; i += 1024)
    (&hbuf[0][0])[i] = __builtin_bit_cast(bf16_t, (unsigned short)0);
  __syncthreads();

  const int lane = threadIdx.x & 31;
  const int w = threadIdx.x >> 5;
  const int j = w & 15, mh = w >> 4;
  const int r16 = lane & 15, half = lane >> 4;

  float creg[2][8];
  float hnew[2][8];
#pragma unroll
  for (int a = 0; a < 2; ++a)
#pragma unroll
    for (int r = 0; r < 8; ++r) creg[a][r] = 0.f;

  for (int si = 0; si < Sn; ++si) {
    const int s = dir ? (Sn - 1 - si) : si;
    const bf16_t* Whp = Wh + opaque_zero();  // force per-step L2 streaming
#pragma unroll
    for (int mtl = 0; mtl < 2; ++mtl) {
      const int m0 = (mh * 2 + mtl) << 4;
      v8f acc[4];
#pragma unroll
      for (int g = 0; g < 4; ++g) {
        const int col = g * Hn + j * 16 + r16;
#pragma unroll
        for (int rr = 0; rr < 8; ++rr) {
          int b = m0 + rr + half * 8;
          acc[g][rr] = xg[(((size_t)b << 10) + s) * 1024 + col];  // bias in GEMM
        }
      }
#pragma unroll
      for (int kc = 0; kc < 8; ++kc) {
        v16bf af = load_frag_a(&hbuf[0][0], 264, m0, kc * 32, lane);
#pragma unroll
        for (int g = 0; g < 4; ++g) {
          v16bf bf = load_frag_b(Whp, Hn, g * Hn + j * 16, kc * 32, lane);
          acc[g] = wmma_bf16(af, bf, acc[g]);
        }
      }
      const int coln = j * 16 + r16;
#pragma unroll
      for (int rr = 0; rr < 8; ++rr) {
        float ig = 1.f / (1.f + __expf(-acc[0][rr]));
        float fg = 1.f / (1.f + __expf(-acc[1][rr]));
        float gg = tanhf(acc[2][rr]);
        float og = 1.f / (1.f + __expf(-acc[3][rr]));
        float c = fg * creg[mtl][rr] + ig * gg;
        creg[mtl][rr] = c;
        float h = og * tanhf(c);
        hnew[mtl][rr] = h;
        int b = m0 + rr + half * 8;
        h_out[(((size_t)b << 10) + s) * En + colofs + coln] = f2bf(h);
      }
    }
    __syncthreads();  // all reads of hbuf done
#pragma unroll
    for (int mtl = 0; mtl < 2; ++mtl) {
      const int m0 = (mh * 2 + mtl) << 4;
      const int coln = j * 16 + r16;
#pragma unroll
      for (int rr = 0; rr < 8; ++rr)
        hbuf[m0 + rr + half * 8][coln] = f2bf(hnew[mtl][rr]);
    }
    __syncthreads();  // h_s visible for next step
  }
}

// ---------------------------------------------------------------------------
// Attention over the batch axis, per (s, head). block=256 (8 waves).
// q/k staged global->LDS by the Tensor Data Mover with LDS row padding
// (64 dw data + 4 dw pad -> 136-element rows matching the frag loader).
__global__ __launch_bounds__(256) void attn_kernel(
    const bf16_t* __restrict__ qkv, bf16_t* __restrict__ o) {
  const int sh = blockIdx.x;
  const int s = sh >> 2, head = sh & 3;
  __shared__ __align__(16) char smem[3 * 17408];
  bf16_t* qs = (bf16_t*)smem;            // [64][136]
  bf16_t* ks = (bf16_t*)(smem + 17408);  // [64][136]
  float*  sc = (float*)(smem + 34816);   // [64][68]
  bf16_t* aa = qs;                       // [64][72]  (reuses q region)
  bf16_t* vt = ks;                       // [128][68] (reuses k region)
  const int QS = 136, SS = 68, AS = 72, VS = 68;

  // TDM DMA: 64 rows x 64 dwords, global row stride S*3E*2 B = 786432 dwords
  if (threadIdx.x < 32) {
    const char* qbase = (const char*)qkv + ((size_t)s * 1536 + head * HDn) * 2;
    tdm_load_2d(qbase, (unsigned)(size_t)(void*)qs, 64, 64, 786432u, 5u, 3u, 1);
    tdm_load_2d(qbase + En * 2, (unsigned)(size_t)(void*)ks, 64, 64, 786432u, 5u, 3u, 1);
    __builtin_amdgcn_s_wait_tensorcnt(0);
  }
  __syncthreads();

  const int lane = threadIdx.x & 31, wv = threadIdx.x >> 5;
  const int r16 = lane & 15, half = lane >> 4;
  const float scale = 0.08838834764831845f;  // 128^-0.5

#pragma unroll
  for (int t = 0; t < 2; ++t) {  // 16 score tiles / 8 waves
    int tile = wv * 2 + t;
    int m0 = (tile >> 2) << 4, n0 = (tile & 3) << 4;
    v8f acc = zero8();
    for (int k0 = 0; k0 < HDn; k0 += 32) {
      v16bf af = load_frag_a(qs, QS, m0, k0, lane);
      v16bf bf = load_frag_b(ks, QS, n0, k0, lane);
      acc = wmma_bf16(af, bf, acc);
    }
    int col = n0 + r16;
#pragma unroll
    for (int rr = 0; rr < 8; ++rr)
      sc[(m0 + rr + half * 8) * SS + col] = acc[rr] * scale;
  }
  __syncthreads();  // q,k dead; scores ready

  if (threadIdx.x < Bn) {  // softmax over m per row l, a into q region
    int l = threadIdx.x;
    float mx = -3e38f;
    for (int m = 0; m < Bn; ++m) mx = fmaxf(mx, sc[l * SS + m]);
    float sum = 0.f;
    for (int m = 0; m < Bn; ++m) sum += __expf(sc[l * SS + m] - mx);
    float inv = 1.f / sum;
    for (int m = 0; m < Bn; ++m) aa[l * AS + m] = f2bf(__expf(sc[l * SS + m] - mx) * inv);
  }
  for (int idx = threadIdx.x; idx < Bn * HDn; idx += 256) {  // vT into k region
    int b = idx >> 7, d = idx & 127;
    vt[d * VS + b] = qkv[(((size_t)b << 10) + s) * 1536 + 2 * En + head * HDn + d];
  }
  __syncthreads();

#pragma unroll
  for (int t = 0; t < 4; ++t) {  // 32 o-tiles / 8 waves; M=64,N=128,K=64
    int tile = wv * 4 + t;
    int m0 = (tile >> 3) << 4, n0 = (tile & 7) << 4;
    v8f acc = zero8();
    for (int k0 = 0; k0 < Bn; k0 += 32) {
      v16bf af = load_frag_a(aa, AS, m0, k0, lane);
      v16bf bf = load_frag_b(vt, VS, n0, k0, lane);
      acc = wmma_bf16(af, bf, acc);
    }
    int col = n0 + r16;
#pragma unroll
    for (int rr = 0; rr < 8; ++rr) {
      int l = m0 + rr + half * 8;
      o[(((size_t)l << 10) + s) * En + head * HDn + col] = f2bf(acc[rr]);
    }
  }
}

// ---------------------------------------------------------------------------
// Viterbi forward: one wave per batch row; lane j tracks tag j.
__global__ __launch_bounds__(32) void viterbi_fwd(
    const float* __restrict__ em, const unsigned char* __restrict__ mask,
    const float* __restrict__ trans, const float* __restrict__ startv,
    const float* __restrict__ endv, unsigned char* __restrict__ hist,
    int* __restrict__ last) {
  const int b = blockIdx.x, j = threadIdx.x;
  const bool act = j < Tn;
  float tcol[Tn];
#pragma unroll
  for (int i = 0; i < Tn; ++i) tcol[i] = act ? trans[i * Tn + j] : 0.f;
  float score = act ? startv[j] + em[((size_t)b << 10) * 32 + j] : -3e38f;
  for (int s = 1; s < Sn; ++s) {
    float e = act ? em[(((size_t)b << 10) + s) * 32 + j] : 0.f;
    float best = -3e38f;
    int arg = 0;
#pragma unroll
    for (int i = 0; i < Tn; ++i) {
      float cand = __shfl(score, i, 32) + tcol[i];
      if (cand > best) { best = cand; arg = i; }
    }
    best += e;
    int m = mask[((size_t)b << 10) + s];
    score = m ? best : score;
    int idx = m ? arg : j;
    if (act) hist[((size_t)s << 11) + (b << 5) + j] = (unsigned char)idx;
  }
  float fin = act ? score + endv[j] : -3e38f;
  float bv = -3e38f;
  int bestj = 0;
#pragma unroll
  for (int i = 0; i < Tn; ++i) {
    float v = __shfl(fin, i, 32);
    if (v > bv) { bv = v; bestj = i; }
  }
  if (j == 0) last[b] = bestj;
}

__global__ __launch_bounds__(64) void viterbi_back(
    const unsigned char* __restrict__ hist, const int* __restrict__ last,
    float* __restrict__ out) {
  int b = threadIdx.x;
  int tag = last[b];
  out[((size_t)b << 10) + (Sn - 1)] = (float)tag;
  for (int s = Sn - 1; s >= 1; --s) {
    tag = hist[((size_t)s << 11) + (b << 5) + tag];
    out[((size_t)b << 10) + (s - 1)] = (float)tag;
  }
}

// ---------------------------------------------------------------------------
extern "C" void kernel_launch(void* const* d_in, const int* in_sizes, int n_in,
                              void* d_out, int out_size, void* d_ws, size_t ws_size,
                              hipStream_t stream) {
  const int*           ids   = (const int*)d_in[0];
  const unsigned char* mask  = (const unsigned char*)d_in[1];
  const float* embed = (const float*)d_in[2];
  const float* wi_f  = (const float*)d_in[3];
  const float* wh_f  = (const float*)d_in[4];
  const float* bi_f  = (const float*)d_in[5];
  const float* bh_f  = (const float*)d_in[6];
  const float* wi_b  = (const float*)d_in[7];
  const float* wh_b  = (const float*)d_in[8];
  const float* bi_b  = (const float*)d_in[9];
  const float* bh_b  = (const float*)d_in[10];
  const float* in_w  = (const float*)d_in[11];
  const float* in_b  = (const float*)d_in[12];
  const float* out_w = (const float*)d_in[13];
  const float* out_b = (const float*)d_in[14];
  const float* lw    = (const float*)d_in[15];
  const float* lb    = (const float*)d_in[16];
  const float* trans = (const float*)d_in[17];
  const float* startv= (const float*)d_in[18];
  const float* endv  = (const float*)d_in[19];

  char* w = (char*)d_ws;
  size_t off = 0;
  auto alloc = [&](size_t bytes) -> void* {
    void* p = w + off;
    off += (bytes + 255) & ~(size_t)255;
    return p;
  };

  const size_t M = (size_t)Bn * Sn;  // 65536
  bf16_t* x_bf    = (bf16_t*)alloc(M * Dn * 2);
  bf16_t* wi_f_bf = (bf16_t*)alloc((size_t)4 * Hn * Dn * 2);
  bf16_t* wh_f_bf = (bf16_t*)alloc((size_t)4 * Hn * Hn * 2);
  bf16_t* wi_b_bf = (bf16_t*)alloc((size_t)4 * Hn * Dn * 2);
  bf16_t* wh_b_bf = (bf16_t*)alloc((size_t)4 * Hn * Hn * 2);
  bf16_t* inw_bf  = (bf16_t*)alloc((size_t)3 * En * En * 2);
  bf16_t* wem_bf  = (bf16_t*)alloc((size_t)32 * En * 2);
  float*  bem     = (float*)alloc(32 * 4);
  float*  xg_f    = (float*)alloc(M * 1024 * 4);
  float*  xg_b    = (float*)alloc(M * 1024 * 4);
  bf16_t* h_bf    = (bf16_t*)alloc(M * En * 2);
  bf16_t* qkv_bf  = (bf16_t*)alloc(M * 3 * En * 2);
  bf16_t* o_bf    = (bf16_t*)alloc(M * En * 2);
  float*  em_f    = (float*)alloc(M * 32 * 4);
  unsigned char* hist = (unsigned char*)alloc((size_t)Sn * Bn * 32);
  int*    last    = (int*)alloc(Bn * 4);

  {
    int n = 4 * Hn * Dn;
    f2bf_kernel<<<(n + 255) / 256, 256, 0, stream>>>(wi_f, wi_f_bf, n);
    f2bf_kernel<<<(n + 255) / 256, 256, 0, stream>>>(wi_b, wi_b_bf, n);
    int nh = 4 * Hn * Hn;
    f2bf_kernel<<<(nh + 255) / 256, 256, 0, stream>>>(wh_f, wh_f_bf, nh);
    f2bf_kernel<<<(nh + 255) / 256, 256, 0, stream>>>(wh_b, wh_b_bf, nh);
    int nw = 3 * En * En;
    f2bf_kernel<<<(nw + 255) / 256, 256, 0, stream>>>(in_w, inw_bf, nw);
  }
  wem_kernel<<<32, 512, 0, stream>>>(lw, lb, out_w, out_b, wem_bf, bem);
  embed_kernel<<<(unsigned)M, 256, 0, stream>>>(ids, embed, x_bf);

  {
    unsigned blocks = (unsigned)((M / 16) * (1024 / 16) / 8);  // 32768
    gemm_bf16_kernel<false><<<blocks, 256, 0, stream>>>(
        x_bf, wi_f_bf, bi_f, bh_f, xg_f, (int)M, 1024, Dn, 1024);
    gemm_bf16_kernel<false><<<blocks, 256, 0, stream>>>(
        x_bf, wi_b_bf, bi_b, bh_b, xg_b, (int)M, 1024, Dn, 1024);
  }

  lstm_kernel<<<2, 1024, 0, stream>>>(xg_f, xg_b, wh_f_bf, wh_b_bf, h_bf);

  {
    unsigned blocks = (unsigned)((M / 16) * (3 * En / 16) / 8);  // 49152
    gemm_bf16_kernel<true><<<blocks, 256, 0, stream>>>(
        h_bf, inw_bf, in_b, nullptr, qkv_bf, (int)M, 3 * En, En, 3 * En);
  }

  attn_kernel<<<Sn * NHn, 256, 0, stream>>>(qkv_bf, o_bf);

  {
    unsigned blocks = (unsigned)((M / 16) * (32 / 16) / 8);  // 1024
    gemm_bf16_kernel<false><<<blocks, 256, 0, stream>>>(
        o_bf, wem_bf, bem, nullptr, em_f, (int)M, 32, En, 32);
  }

  viterbi_fwd<<<Bn, 32, 0, stream>>>(em_f, mask, trans, startv, endv, hist, last);
  viterbi_back<<<1, 64, 0, stream>>>(hist, last, (float*)d_out);
}
